// AttentionBlock_18571438588076
// MI455X (gfx1250) — matrix operand order repaired and checked by reading, not compiled, not run on hardware
//
#include <hip/hip_runtime.h>

// ---------------------------------------------------------------------------
// AttentionBlock for MI455X (gfx1250): GroupNorm -> QKV -> MHA -> proj + res
// All matmuls via v_wmma_f32_16x16x32_bf16 (fp32 accumulate).
// Register blocking: 32x64 tiles per wave (8 wmma accumulators) so each
// loaded fragment feeds >=2 WMMAs (compute-bound workload; L2-resident).
// ---------------------------------------------------------------------------

typedef __attribute__((ext_vector_type(16))) __bf16        v16bf;
typedef __attribute__((ext_vector_type(8)))  float         v8f;
typedef __attribute__((ext_vector_type(4)))  unsigned int  u32x4;

union Frag { u32x4 u[2]; v16bf v; };

#define NB   8
#define NC   512
#define NHW  1024
#define NH   8
#define HD   64
#define SEQ  1024

__device__ __forceinline__ unsigned short f2bf(float f) {
  unsigned int u = __float_as_uint(f);
  u += 0x7FFFu + ((u >> 16) & 1u);          // round-to-nearest-even
  return (unsigned short)(u >> 16);
}

__device__ __forceinline__ v8f wmma_bf16(v16bf a, v16bf b, v8f c) {
  // D = A(16x32 bf16) * B(32x16 bf16) + C(16x16 f32)
  return __builtin_amdgcn_wmma_f32_16x16x32_bf16(
      false, a, false, b, (short)0, c, false, false);
}

// A-fragment (16x32, row-major src): lane m holds row; k runs of 8 at
// k0+half*8 and k0+16+half*8 (ISA 16-bit A layout).
__device__ __forceinline__ Frag load_a(const unsigned short* base, int ld,
                                       int m, int half, int k0) {
  Frag f;
  const unsigned short* row = base + (size_t)m * ld + k0;
  f.u[0] = *(const u32x4*)(row + half * 8);
  f.u[1] = *(const u32x4*)(row + 16 + half * 8);
  return f;
}

// B-fragment (32x16): B[k][n] = W[n][k] with W row-major (N x K).
// lane n holds a column; k runs of 8 at k0+half*8 and k0+16+half*8 (matches A).
__device__ __forceinline__ Frag load_b(const unsigned short* base, int ld,
                                       int n, int half, int k0) {
  Frag f;
  const unsigned short* row = base + (size_t)n * ld + k0;
  f.u[0] = *(const u32x4*)(row + half * 8);
  f.u[1] = *(const u32x4*)(row + 16 + half * 8);
  return f;
}

// ---------------------------------------------------------------------------
// 1) GroupNorm over (C/G, H, W) = (16, 32, 32); write xn transposed to
//    (B*HW, C) bf16 so GEMM A-rows are contiguous.
// ---------------------------------------------------------------------------
__global__ __launch_bounds__(256) void gn_kernel(const float* __restrict__ x,
                                                 const float* __restrict__ gw,
                                                 const float* __restrict__ gb,
                                                 unsigned short* __restrict__ xn) {
  const int bg = blockIdx.x;           // 0..255
  const int b = bg >> 5, g = bg & 31;
  const float* xp = x + ((size_t)b * NC + g * 16) * NHW;
  const int tid = threadIdx.x;

  float s = 0.f, s2 = 0.f;
  for (int i = tid; i < 16 * NHW; i += 256) {
    float v = xp[i];
    s += v; s2 += v * v;
  }
  __shared__ float red[256];
  red[tid] = s; __syncthreads();
  for (int off = 128; off > 0; off >>= 1) {
    if (tid < off) red[tid] += red[tid + off];
    __syncthreads();
  }
  float tot = red[0]; __syncthreads();
  red[tid] = s2; __syncthreads();
  for (int off = 128; off > 0; off >>= 1) {
    if (tid < off) red[tid] += red[tid + off];
    __syncthreads();
  }
  float tot2 = red[0];

  const float invN = 1.0f / 16384.0f;
  float mean = tot * invN;
  float var  = tot2 * invN - mean * mean;
  float inv  = rsqrtf(var + 1e-5f);

  for (int i = tid; i < 16 * NHW; i += 256) {
    int cl = i >> 10, hw = i & 1023;
    int c = g * 16 + cl;
    float v = (xp[i] - mean) * inv * gw[c] + gb[c];
    xn[((size_t)(b * NHW) + hw) * NC + c] = f2bf(v);
  }
}

// ---------------------------------------------------------------------------
// 2) fp32 -> bf16 weight conversion (layout preserved, N x K row-major)
// ---------------------------------------------------------------------------
__global__ __launch_bounds__(256) void cvt_bf16_kernel(const float* __restrict__ src,
                                                       unsigned short* __restrict__ dst,
                                                       int n) {
  int i = blockIdx.x * 256 + threadIdx.x;
  if (i < n) dst[i] = f2bf(src[i]);
}

// ---------------------------------------------------------------------------
// 3) QKV GEMM: (8192 x 512) x (512 x 1536). Wave computes a 32x64 tile
//    (2 M-tiles x 4 N-tiles). Epilogue: +bias, scatter to Q (pre-scaled
//    1/sqrt(64)), K, V^T (d-major so PV B-fragments are contiguous).
// ---------------------------------------------------------------------------
__global__ __launch_bounds__(128) void qkv_gemm_kernel(
    const unsigned short* __restrict__ Xn, const unsigned short* __restrict__ Wq,
    const float* __restrict__ bias, unsigned short* __restrict__ Q,
    unsigned short* __restrict__ K, unsigned short* __restrict__ VT) {
  const int lane = threadIdx.x;
  const int m0 = blockIdx.x * 32;
  const int n0 = blockIdx.y * 256 + threadIdx.y * 64;
  const int ml = lane & 15, half = lane >> 4;

  v8f acc[2][4];
#pragma unroll
  for (int rt = 0; rt < 2; ++rt)
#pragma unroll
    for (int t = 0; t < 4; ++t) acc[rt][t] = (v8f){0,0,0,0,0,0,0,0};

  for (int k0 = 0; k0 < NC; k0 += 32) {
    Frag a[2];
#pragma unroll
    for (int rt = 0; rt < 2; ++rt)
      a[rt] = load_a(Xn, NC, m0 + rt * 16 + ml, half, k0);
    __builtin_prefetch(Xn + (size_t)(m0 + ml) * NC + k0 + 32, 0, 1);
#pragma unroll
    for (int t = 0; t < 4; ++t) {
      Frag bf = load_b(Wq, NC, n0 + t * 16 + ml, half, k0);
#pragma unroll
      for (int rt = 0; rt < 2; ++rt)
        acc[rt][t] = wmma_bf16(a[rt].v, bf.v, acc[rt][t]);
    }
  }

  const float scale = 0.125f;  // 64^-0.5
#pragma unroll
  for (int t = 0; t < 4; ++t) {
    int n = n0 + t * 16 + ml;
    float bv = bias[n];
    int which = n >> 9;        // 0=Q 1=K 2=V
    int c = n & 511, h = c >> 6, d = c & 63;
#pragma unroll
    for (int rt = 0; rt < 2; ++rt)
#pragma unroll
      for (int r = 0; r < 8; ++r) {
        int mm = m0 + rt * 16 + r + 8 * half;
        int b = mm >> 10, hw = mm & 1023;
        size_t bh = (size_t)(b * NH + h);
        float v = acc[rt][t][r] + bv;
        if (which == 0)      Q[(bh * SEQ + hw) * HD + d] = f2bf(v * scale);
        else if (which == 1) K[(bh * SEQ + hw) * HD + d] = f2bf(v);
        else                 VT[(bh * HD + d) * SEQ + hw] = f2bf(v);
      }
  }
}

// ---------------------------------------------------------------------------
// 4) Flash attention: one wave per (b,h, 32-row q block). Online softmax;
//    P tiles bounced through LDS to convert C/D layout -> A layout.
//    K/V fragments are reused across both row-tiles (16 wmma / 16 b128).
// ---------------------------------------------------------------------------
__global__ __launch_bounds__(32) void attn_kernel(
    const unsigned short* __restrict__ Q, const unsigned short* __restrict__ K,
    const unsigned short* __restrict__ VT, unsigned short* __restrict__ Out) {
  const int lane = threadIdx.x;
  const int qb = blockIdx.x;            // 0..31 (32-row q block)
  const int bh = blockIdx.y;            // 0..63
  const int b = bh >> 3, h = bh & 7;
  const unsigned short* Qp = Q  + (size_t)bh * SEQ * HD;
  const unsigned short* Kp = K  + (size_t)bh * SEQ * HD;
  const unsigned short* Vp = VT + (size_t)bh * HD * SEQ;
  const int ml = lane & 15, half = lane >> 4;
  const int q0 = qb * 32;

  Frag qa[2][2];
#pragma unroll
  for (int rt = 0; rt < 2; ++rt)
#pragma unroll
    for (int d = 0; d < 2; ++d)
      qa[rt][d] = load_a(Qp, HD, q0 + rt * 16 + ml, half, d * 32);

  v8f acc[2][4];
#pragma unroll
  for (int rt = 0; rt < 2; ++rt)
#pragma unroll
    for (int t = 0; t < 4; ++t) acc[rt][t] = (v8f){0,0,0,0,0,0,0,0};
  float mrow[2][8], lrow[2][8];
#pragma unroll
  for (int rt = 0; rt < 2; ++rt)
#pragma unroll
    for (int r = 0; r < 8; ++r) { mrow[rt][r] = -3.0e38f; lrow[rt][r] = 0.f; }

  __shared__ __align__(16) unsigned short Pt[2][16][32];   // 2 KB

  for (int kb = 0; kb < SEQ; kb += 32) {
    // --- S tiles (2 x 16 q rows x 32 keys); K fragment reused across rt ---
    v8f s[2][2];
#pragma unroll
    for (int rt = 0; rt < 2; ++rt)
#pragma unroll
      for (int nt = 0; nt < 2; ++nt) s[rt][nt] = (v8f){0,0,0,0,0,0,0,0};
#pragma unroll
    for (int nt = 0; nt < 2; ++nt)
#pragma unroll
      for (int d = 0; d < 2; ++d) {
        Frag kf = load_b(Kp, HD, kb + nt * 16 + ml, half, d * 32);
#pragma unroll
        for (int rt = 0; rt < 2; ++rt)
          s[rt][nt] = wmma_bf16(qa[rt][d].v, kf.v, s[rt][nt]);
      }

    // --- online softmax per row-tile (row stats live in 16-lane halves) ---
#pragma unroll
    for (int rt = 0; rt < 2; ++rt) {
      float tmax[8];
#pragma unroll
      for (int r = 0; r < 8; ++r) tmax[r] = fmaxf(s[rt][0][r], s[rt][1][r]);
#pragma unroll
      for (int off = 1; off < 16; off <<= 1)
#pragma unroll
        for (int r = 0; r < 8; ++r)
          tmax[r] = fmaxf(tmax[r], __shfl_xor(tmax[r], off, 32));

      float alpha[8];
#pragma unroll
      for (int r = 0; r < 8; ++r) {
        float mn = fmaxf(mrow[rt][r], tmax[r]);
        alpha[r] = __expf(mrow[rt][r] - mn);
        mrow[rt][r] = mn;
      }
      float rsum[8];
#pragma unroll
      for (int r = 0; r < 8; ++r) rsum[r] = 0.f;
#pragma unroll
      for (int nt = 0; nt < 2; ++nt)
#pragma unroll
        for (int r = 0; r < 8; ++r) {
          float p = __expf(s[rt][nt][r] - mrow[rt][r]);
          s[rt][nt][r] = p;
          rsum[r] += p;
        }
#pragma unroll
      for (int off = 1; off < 16; off <<= 1)
#pragma unroll
        for (int r = 0; r < 8; ++r) rsum[r] += __shfl_xor(rsum[r], off, 32);
#pragma unroll
      for (int r = 0; r < 8; ++r)
        lrow[rt][r] = lrow[rt][r] * alpha[r] + rsum[r];
#pragma unroll
      for (int t = 0; t < 4; ++t)
#pragma unroll
        for (int r = 0; r < 8; ++r) acc[rt][t][r] *= alpha[r];

      // P: C/D layout -> LDS (bf16)
#pragma unroll
      for (int nt = 0; nt < 2; ++nt)
#pragma unroll
        for (int r = 0; r < 8; ++r)
          Pt[rt][r + 8 * half][nt * 16 + ml] = f2bf(s[rt][nt][r]);
    }
    __syncthreads();

    Frag pa[2];
#pragma unroll
    for (int rt = 0; rt < 2; ++rt) {
      pa[rt].u[0] = *(const u32x4*)(&Pt[rt][ml][half * 8]);
      pa[rt].u[1] = *(const u32x4*)(&Pt[rt][ml][16 + half * 8]);
    }

    // --- O += P * V ; V fragment reused across rt ---
#pragma unroll
    for (int t = 0; t < 4; ++t) {
      Frag vf = load_b(Vp, SEQ, t * 16 + ml, half, kb);
#pragma unroll
      for (int rt = 0; rt < 2; ++rt)
        acc[rt][t] = wmma_bf16(pa[rt].v, vf.v, acc[rt][t]);
    }
    __syncthreads();
  }

  // epilogue: normalize and store to (8192 x 512) bf16, col = h*64 + d
#pragma unroll
  for (int rt = 0; rt < 2; ++rt)
#pragma unroll
    for (int t = 0; t < 4; ++t)
#pragma unroll
      for (int r = 0; r < 8; ++r) {
        int row = q0 + rt * 16 + r + 8 * half;
        int col = h * HD + t * 16 + ml;
        float o = acc[rt][t][r] / lrow[rt][r];
        Out[((size_t)(b * NHW) + row) * NC + col] = f2bf(o);
      }
}

// ---------------------------------------------------------------------------
// 5) proj GEMM (8192x512 x 512x512) + bias + residual, fp32 out (B,C,H,W)
// ---------------------------------------------------------------------------
__global__ __launch_bounds__(128) void proj_gemm_kernel(
    const unsigned short* __restrict__ A, const unsigned short* __restrict__ Wp,
    const float* __restrict__ bias, const float* __restrict__ x,
    float* __restrict__ out) {
  const int lane = threadIdx.x;
  const int m0 = blockIdx.x * 32;
  const int n0 = blockIdx.y * 256 + threadIdx.y * 64;
  const int ml = lane & 15, half = lane >> 4;

  v8f acc[2][4];
#pragma unroll
  for (int rt = 0; rt < 2; ++rt)
#pragma unroll
    for (int t = 0; t < 4; ++t) acc[rt][t] = (v8f){0,0,0,0,0,0,0,0};

  for (int k0 = 0; k0 < NC; k0 += 32) {
    Frag a[2];
#pragma unroll
    for (int rt = 0; rt < 2; ++rt)
      a[rt] = load_a(A, NC, m0 + rt * 16 + ml, half, k0);
#pragma unroll
    for (int t = 0; t < 4; ++t) {
      Frag bf = load_b(Wp, NC, n0 + t * 16 + ml, half, k0);
#pragma unroll
      for (int rt = 0; rt < 2; ++rt)
        acc[rt][t] = wmma_bf16(a[rt].v, bf.v, acc[rt][t]);
    }
  }

#pragma unroll
  for (int t = 0; t < 4; ++t) {
    int n = n0 + t * 16 + ml;
    float bv = bias[n];
#pragma unroll
    for (int rt = 0; rt < 2; ++rt)
#pragma unroll
      for (int r = 0; r < 8; ++r) {
        int mm = m0 + rt * 16 + r + 8 * half;
        int b = mm >> 10, hw = mm & 1023;
        size_t oi = ((size_t)b * NC + n) * NHW + hw;
        out[oi] = acc[rt][t][r] + bv + x[oi];
      }
  }
}

// ---------------------------------------------------------------------------
extern "C" void kernel_launch(void* const* d_in, const int* in_sizes, int n_in,
                              void* d_out, int out_size, void* d_ws, size_t ws_size,
                              hipStream_t stream) {
  const float* x      = (const float*)d_in[0];
  const float* gn_w   = (const float*)d_in[1];
  const float* gn_b   = (const float*)d_in[2];
  const float* qkv_w  = (const float*)d_in[3];
  const float* qkv_b  = (const float*)d_in[4];
  const float* proj_w = (const float*)d_in[5];
  const float* proj_b = (const float*)d_in[6];
  float* out = (float*)d_out;

  char* ws = (char*)d_ws;
  unsigned short* xn  = (unsigned short*)ws; ws += (size_t)NB * NHW * NC * 2;
  unsigned short* wq  = (unsigned short*)ws; ws += (size_t)3 * NC * NC * 2;
  unsigned short* wp  = (unsigned short*)ws; ws += (size_t)NC * NC * 2;
  unsigned short* Q   = (unsigned short*)ws; ws += (size_t)NB * NH * SEQ * HD * 2;
  unsigned short* K   = (unsigned short*)ws; ws += (size_t)NB * NH * SEQ * HD * 2;
  unsigned short* VT  = (unsigned short*)ws; ws += (size_t)NB * NH * HD * SEQ * 2;
  unsigned short* att = (unsigned short*)ws; ws += (size_t)NB * NHW * NC * 2;

  // 1) GroupNorm -> transposed bf16 activations
  gn_kernel<<<dim3(NB * 32), dim3(256), 0, stream>>>(x, gn_w, gn_b, xn);

  // 2) weight conversions
  {
    int n1 = 3 * NC * NC;
    cvt_bf16_kernel<<<dim3((n1 + 255) / 256), dim3(256), 0, stream>>>(qkv_w, wq, n1);
    int n2 = NC * NC;
    cvt_bf16_kernel<<<dim3((n2 + 255) / 256), dim3(256), 0, stream>>>(proj_w, wp, n2);
  }

  // 3) QKV GEMM -> Q (scaled), K, V^T
  qkv_gemm_kernel<<<dim3(NB * NHW / 32, (3 * NC) / 256), dim3(32, 4), 0, stream>>>(
      xn, wq, qkv_b, Q, K, VT);

  // 4) attention
  attn_kernel<<<dim3(SEQ / 32, NB * NH), dim3(32), 0, stream>>>(Q, K, VT, att);

  // 5) projection + residual
  proj_gemm_kernel<<<dim3(NB * NHW / 32, NC / 256), dim3(32, 4), 0, stream>>>(
      att, wp, proj_b, x, out);
}
